// Block_72739566125973
// MI455X (gfx1250) — matrix operand order repaired and checked by reading, compiled, not run on hardware
//
#include <hip/hip_runtime.h>
#include <math.h>

// ---------------------------------------------------------------------------
// Transformer block (pre-LN GPT style) for MI455X / gfx1250, wave32 + WMMA.
// Matrix math: v_wmma_f32_16x16x32_bf16 (f32 accumulate).
// Reductions:  DPP8 + permlane16 (pure VALU).
// Staging:     global_load_async_to_lds_b128 (ASYNCcnt) double buffering.
// ---------------------------------------------------------------------------

typedef float  v8f   __attribute__((ext_vector_type(8)));
typedef __bf16 v16bf __attribute__((ext_vector_type(16)));

#define DIM_D   1024
#define DIM_B   2
#define DIM_T   2048
#define DIM_NH  16
#define DIM_HD  64
#define DIM_FF  4096
#define DIM_M   (DIM_B * DIM_T)   // 4096 tokens

static __device__ __forceinline__ unsigned short f2bf(float f) {
    unsigned int u = __float_as_uint(f);
    u += 0x7fffu + ((u >> 16) & 1u);          // round-to-nearest-even
    return (unsigned short)(u >> 16);
}

// ---- async global->LDS copy (CDNA5; 16B per lane, tracked by ASYNCcnt) ----
static __device__ __forceinline__ void async_copy16(const void* lds_dst,
                                                    const void* gsrc) {
    unsigned lds_off = (unsigned)(unsigned long long)lds_dst;  // addr[31:0] = LDS offset
    asm volatile("global_load_async_to_lds_b128 %0, %1, off"
                 :: "v"(lds_off), "v"((unsigned long long)gsrc)
                 : "memory");
}
static __device__ __forceinline__ void wait_async0() {
    asm volatile("s_wait_asynccnt 0x0" ::: "memory");
}

// ---- DPP / permlane lane-reduction helpers (wave32) ------------------------
constexpr int dpp8_xor_sel(int k) {
    int v = 0;
    for (int i = 0; i < 8; i++) v |= ((i ^ k) & 7) << (3 * i);
    return v;
}
constexpr int DPP8_X1 = dpp8_xor_sel(1);
constexpr int DPP8_X2 = dpp8_xor_sel(2);
constexpr int DPP8_X4 = dpp8_xor_sel(4);

#define DPP8F(x, sel) \
    __int_as_float(__builtin_amdgcn_mov_dpp8(__float_as_int(x), (sel)))
// xor-8 within each 16-lane row
#define PL16_X8(x)                                                        \
    __int_as_float(__builtin_amdgcn_permlane16(                           \
        __float_as_int(x), __float_as_int(x), (int)0xFEDCBA98,            \
        (int)0x76543210, false, false))
// swap the two 16-lane rows of the wave
#define PLX16(x)                                                          \
    __int_as_float(__builtin_amdgcn_permlanex16(                          \
        __float_as_int(x), __float_as_int(x), (int)0x76543210,            \
        (int)0xFEDCBA98, false, false))

static __device__ __forceinline__ float row16_max(float x) {
    x = fmaxf(x, DPP8F(x, DPP8_X1));
    x = fmaxf(x, DPP8F(x, DPP8_X2));
    x = fmaxf(x, DPP8F(x, DPP8_X4));
    x = fmaxf(x, PL16_X8(x));
    return x;
}
static __device__ __forceinline__ float row16_sum(float x) {
    x += DPP8F(x, DPP8_X1);
    x += DPP8F(x, DPP8_X2);
    x += DPP8F(x, DPP8_X4);
    x += PL16_X8(x);
    return x;
}
static __device__ __forceinline__ float wave32_sum(float x) {
    x = row16_sum(x);
    x += PLX16(x);
    return x;
}

// ---- WMMA fragment helpers -------------------------------------------------
// A-matrix layout: two 16B chunks (adjacent at every call: contiguous k).
static __device__ __forceinline__ v16bf frag2(const unsigned short* p0,
                                              const unsigned short* p1) {
    union { uint4 u[2]; v16bf v; } t;
    t.u[0] = *(const uint4*)p0;
    t.u[1] = *(const uint4*)p1;
    return t.v;
}
// B-matrix layout: one 32B contiguous chunk.
static __device__ __forceinline__ v16bf frag1(const unsigned short* p) {
    union { uint4 u[2]; v16bf v; } t;
    t.u[0] = *(const uint4*)p;
    t.u[1] = *(const uint4*)(p + 8);
    return t.v;
}

static __device__ __forceinline__ v8f wmma_bf16(v16bf a, v16bf b, v8f c) {
    return __builtin_amdgcn_wmma_f32_16x16x32_bf16(false, a, false, b,
                                                   (short)0, c, false, false);
}

static __device__ __forceinline__ float gelu_exact(float z) {
    return 0.5f * z * (1.0f + erff(z * 0.70710678118654752f));
}

// ---------------------------------------------------------------------------
// Weight transpose + f32->bf16 convert:  in[K,N] (row major) -> out[N,K] bf16
// ---------------------------------------------------------------------------
__global__ __launch_bounds__(256)
void tconv_kernel(const float* __restrict__ in, unsigned short* __restrict__ out,
                  int K, int N) {
    __shared__ float tile[32][33];
    int n0 = blockIdx.x * 32, k0 = blockIdx.y * 32;
    int tx = threadIdx.x, ty = threadIdx.y;   // 32 x 8
    #pragma unroll
    for (int i = 0; i < 32; i += 8)
        tile[ty + i][tx] = in[(size_t)(k0 + ty + i) * N + n0 + tx];
    __syncthreads();
    #pragma unroll
    for (int i = 0; i < 32; i += 8)
        out[(size_t)(n0 + ty + i) * K + k0 + tx] = f2bf(tile[tx][ty + i]);
}

// ---------------------------------------------------------------------------
// LayerNorm over D=1024, one row per block (256 threads, 4 floats each).
// ---------------------------------------------------------------------------
__global__ __launch_bounds__(256)
void ln_kernel(const float* __restrict__ x, const float* __restrict__ g,
               const float* __restrict__ b, unsigned short* __restrict__ out) {
    int row = blockIdx.x;
    int t = threadIdx.x;
    const float4 xv = ((const float4*)(x + (size_t)row * DIM_D))[t];
    float s  = xv.x + xv.y + xv.z + xv.w;
    float s2 = xv.x * xv.x + xv.y * xv.y + xv.z * xv.z + xv.w * xv.w;
    s  = wave32_sum(s);
    s2 = wave32_sum(s2);
    __shared__ float rs[8], rs2[8];
    __shared__ float smu, srstd;
    int w = t >> 5, l = t & 31;
    if (l == 0) { rs[w] = s; rs2[w] = s2; }
    __syncthreads();
    if (t == 0) {
        float S = 0.f, S2 = 0.f;
        #pragma unroll
        for (int i = 0; i < 8; i++) { S += rs[i]; S2 += rs2[i]; }
        float mu  = S * (1.0f / DIM_D);
        float var = S2 * (1.0f / DIM_D) - mu * mu;
        smu = mu;
        srstd = rsqrtf(var + 1e-5f);
    }
    __syncthreads();
    float mu = smu, r = srstd;
    int c = t * 4;
    union { unsigned short u[4]; uint2 d; } o;
    o.u[0] = f2bf((xv.x - mu) * r * g[c + 0] + b[c + 0]);
    o.u[1] = f2bf((xv.y - mu) * r * g[c + 1] + b[c + 1]);
    o.u[2] = f2bf((xv.z - mu) * r * g[c + 2] + b[c + 2]);
    o.u[3] = f2bf((xv.w - mu) * r * g[c + 3] + b[c + 3]);
    *(uint2*)(out + (size_t)row * DIM_D + c) = o.d;
}

// ---------------------------------------------------------------------------
// Tiled bf16 WMMA GEMM:  C[M,N] = A[M,K] x Bt[N,K]^T    (f32 accumulate)
// Block tile 128x64, BK=64, 256 threads = 8 waves (4x2), wave tile 32x32.
// LDS double-buffered via async global->LDS copies; one barrier per BK step,
// 8 WMMAs per step; next tile streams in while WMMA chain runs.
// MODE 0: scatter to q/k/v [B,NH,T,HD] bf16, q scaled by 1/sqrt(HD)
// MODE 1: + resid           -> f32 out
// MODE 2: + bias, GELU      -> bf16 out
// MODE 3: + bias + resid    -> f32 out
// ---------------------------------------------------------------------------
template <int MODE>
__global__ __launch_bounds__(256)
void gemm_bf16_kernel(const unsigned short* __restrict__ A,
                      const unsigned short* __restrict__ Bt,
                      const float* __restrict__ bias,
                      const float* __restrict__ resid,
                      float* __restrict__ outF,
                      unsigned short* __restrict__ outB,
                      unsigned short* __restrict__ qg,
                      unsigned short* __restrict__ kg,
                      unsigned short* __restrict__ vg,
                      int M, int N, int K) {
    __shared__ __align__(16) unsigned short sA[2][128 * 72];  // 64 cols, pad 72
    __shared__ __align__(16) unsigned short sB[2][64 * 72];

    const int tid = threadIdx.x;
    const int l = tid & 31, w = tid >> 5;
    const int wm = w >> 1, wn = w & 1;
    const int hh = l >> 4, ln = l & 15;
    const int m0 = blockIdx.y * 128, n0 = blockIdx.x * 64;

    v8f acc[2][2];
    #pragma unroll
    for (int i = 0; i < 2; i++)
        #pragma unroll
        for (int j = 0; j < 2; j++)
            acc[i][j] = (v8f){0.f, 0.f, 0.f, 0.f, 0.f, 0.f, 0.f, 0.f};

    const int arow = tid >> 1, ahalf = tid & 1;     // A staging: 64B / thread
    const int bn = tid >> 2,  bq = tid & 3;         // B staging: 32B / thread
    const unsigned short* Aptr = A + (size_t)(m0 + arow) * K + ahalf * 32;
    const unsigned short* Bptr = Bt + (size_t)(n0 + bn) * K + bq * 16;
    unsigned short* const sAw = &sA[0][0] + arow * 72 + ahalf * 32;
    unsigned short* const sBw = &sB[0][0] + bn * 72 + bq * 16;
    const size_t abufsz = 128 * 72, bbufsz = 64 * 72;

    auto issueTile = [&](int buf, int k0) {
        #pragma unroll
        for (int j = 0; j < 4; j++)
            async_copy16(sAw + buf * abufsz + j * 8, Aptr + k0 + j * 8);
        #pragma unroll
        for (int j = 0; j < 2; j++)
            async_copy16(sBw + buf * bbufsz + j * 8, Bptr + k0 + j * 8);
    };

    issueTile(0, 0);
    wait_async0();
    __syncthreads();

    int buf = 0;
    for (int k0 = 0; k0 < K; k0 += 64) {
        const bool more = (k0 + 64) < K;
        if (more) issueTile(buf ^ 1, k0 + 64);   // streams during WMMA chain

        const unsigned short* pa = &sA[buf][0];
        const unsigned short* pb = &sB[buf][0];
        #pragma unroll
        for (int kc = 0; kc < 2; kc++) {
            v16bf af[2], bfr[2];
            #pragma unroll
            for (int rt = 0; rt < 2; rt++) {
                int r = wm * 32 + rt * 16 + ln;
                af[rt] = frag2(pa + r * 72 + kc * 32 + hh * 16,
                               pa + r * 72 + kc * 32 + hh * 16 + 8);
            }
            #pragma unroll
            for (int ct = 0; ct < 2; ct++) {
                int n = wn * 32 + ct * 16 + ln;
                bfr[ct] = frag1(pb + n * 72 + kc * 32 + hh * 16);
            }
            #pragma unroll
            for (int rt = 0; rt < 2; rt++)
                #pragma unroll
                for (int ct = 0; ct < 2; ct++)
                    acc[rt][ct] = wmma_bf16(af[rt], bfr[ct], acc[rt][ct]);
        }

        if (more) {
            wait_async0();
            __syncthreads();
            buf ^= 1;
        }
    }

    // Epilogue: C layout — row = v + 8*(lane>>4), col = lane&15
    #pragma unroll
    for (int rt = 0; rt < 2; rt++) {
        #pragma unroll
        for (int ct = 0; ct < 2; ct++) {
            #pragma unroll
            for (int v = 0; v < 8; v++) {
                float val = acc[rt][ct][v];
                int m = m0 + wm * 32 + rt * 16 + v + 8 * hh;
                int n = n0 + wn * 32 + ct * 16 + ln;
                if (MODE == 0) {
                    int bb = m >> 11, t = m & (DIM_T - 1);
                    if (n < DIM_D) {
                        int h = n >> 6, hd = n & 63;
                        qg[(((size_t)(bb * DIM_NH + h)) * DIM_T + t) * DIM_HD + hd] =
                            f2bf(val * 0.125f);                  // 1/sqrt(64)
                    } else if (n < 2 * DIM_D) {
                        int n2 = n - DIM_D;
                        int h = n2 >> 6, hd = n2 & 63;
                        kg[(((size_t)(bb * DIM_NH + h)) * DIM_T + t) * DIM_HD + hd] = f2bf(val);
                    } else {
                        int n2 = n - 2 * DIM_D;
                        int h = n2 >> 6, hd = n2 & 63;
                        vg[(((size_t)(bb * DIM_NH + h)) * DIM_T + t) * DIM_HD + hd] = f2bf(val);
                    }
                } else if (MODE == 1) {
                    outF[(size_t)m * N + n] = val + resid[(size_t)m * N + n];
                } else if (MODE == 2) {
                    outB[(size_t)m * N + n] = f2bf(gelu_exact(val + bias[n]));
                } else {
                    outF[(size_t)m * N + n] = val + bias[n] + resid[(size_t)m * N + n];
                }
            }
        }
    }
}

// ---------------------------------------------------------------------------
// Flash-style causal attention. Grid: (T/64, B*NH). 128 threads = 4 waves,
// wave w owns 16 query rows. 32 keys per step; S and P*V via WMMA.
// K tiles staged with async global->LDS; K/V double-buffered -> one barrier
// per key block. P handoff is wave-private LDS (same-wave DS ops in-order).
// ---------------------------------------------------------------------------
__global__ __launch_bounds__(128)
void attn_kernel(const unsigned short* __restrict__ qg,
                 const unsigned short* __restrict__ kg,
                 const unsigned short* __restrict__ vg,
                 unsigned short* __restrict__ out) {
    __shared__ __align__(16) unsigned short sK[2][32 * 72];     // [key][hd]
    __shared__ __align__(16) unsigned short sV[2][64 * 40];     // [hd][key]
    __shared__ __align__(16) unsigned short sP[4 * 16 * 40];    // per-wave P

    const int bh = blockIdx.y;                    // b*NH + h
    const int q0 = blockIdx.x * 64;
    const int tid = threadIdx.x;
    const int l = tid & 31, w = tid >> 5;
    const int hh = l >> 4, ln = l & 15;
    const size_t base = (size_t)bh * DIM_T * DIM_HD;
    const int qr0 = q0 + w * 16;

    // Query fragments for this wave's 16 rows (K = 64 -> two fragments)
    v16bf qf[2];
    {
        const unsigned short* qp = qg + base + (size_t)(qr0 + ln) * DIM_HD;
        qf[0] = frag2(qp + hh * 16,      qp + hh * 16 + 8);
        qf[1] = frag2(qp + 32 + hh * 16, qp + 40 + hh * 16);
    }

    float mst[8], lst[8];
    v8f o[4];
    #pragma unroll
    for (int i = 0; i < 8; i++) { mst[i] = -3.0e38f; lst[i] = 0.f; }
    #pragma unroll
    for (int i = 0; i < 4; i++) o[i] = (v8f){0.f,0.f,0.f,0.f,0.f,0.f,0.f,0.f};

    const int skey = tid >> 2, sq = tid & 3;            // K staging
    const int vkey = tid & 31, vhb = (tid >> 5) * 16;   // V staging

    int buf = 0;
    for (int j0 = 0; j0 <= q0 + 63; j0 += 32) {
        {   // stage K tile [32 keys][64] row-major via async copies
            const unsigned short* kp = kg + base + (size_t)(j0 + skey) * DIM_HD + sq * 16;
            unsigned short* kd = &sK[buf][skey * 72 + sq * 16];
            async_copy16(kd, kp);
            async_copy16(kd + 8, kp + 8);
        }
        {   // stage V tile transposed: sV[hd][key]
            unsigned short tmp[16];
            *(uint4*)(tmp)     = *(const uint4*)(vg + base + (size_t)(j0 + vkey) * DIM_HD + vhb);
            *(uint4*)(tmp + 8) = *(const uint4*)(vg + base + (size_t)(j0 + vkey) * DIM_HD + vhb + 8);
            #pragma unroll
            for (int i = 0; i < 16; i++) sV[buf][(vhb + i) * 40 + vkey] = tmp[i];
        }
        wait_async0();
        __syncthreads();

        // S = q x k^T  (two 16-key column tiles, K = 64)
        v8f s[2];
        #pragma unroll
        for (int ct = 0; ct < 2; ct++) {
            s[ct] = (v8f){0.f,0.f,0.f,0.f,0.f,0.f,0.f,0.f};
            int key = ct * 16 + ln;
            v16bf k0f = frag1(&sK[buf][key * 72 + hh * 16]);
            v16bf k1f = frag1(&sK[buf][key * 72 + 32 + hh * 16]);
            s[ct] = wmma_bf16(qf[0], k0f, s[ct]);
            s[ct] = wmma_bf16(qf[1], k1f, s[ct]);
        }

        // causal mask + online softmax (row = v + 8*hh lives in one 16-lane row)
        unsigned short* pbp = sP + w * 640;
        #pragma unroll
        for (int v = 0; v < 8; v++) {
            int trow = qr0 + v + 8 * hh;
            float s0 = (j0 + ln      > trow) ? -3.0e38f : s[0][v];
            float s1 = (j0 + 16 + ln > trow) ? -3.0e38f : s[1][v];
            float mx = row16_max(fmaxf(s0, s1));
            float mnew = fmaxf(mst[v], mx);
            float p0 = __expf(s0 - mnew);
            float p1 = __expf(s1 - mnew);
            float rsum = row16_sum(p0 + p1);
            float sc = __expf(mst[v] - mnew);
            lst[v] = lst[v] * sc + rsum;
            mst[v] = mnew;
            #pragma unroll
            for (int nt = 0; nt < 4; nt++) o[nt][v] *= sc;
            int prow = v + 8 * hh;
            pbp[prow * 40 + ln]      = f2bf(p0);
            pbp[prow * 40 + 16 + ln] = f2bf(p1);
        }
        asm volatile("" ::: "memory");   // keep P stores before P loads

        // O += P x V   (K = 32 keys, four 16-wide hd tiles)
        v16bf pf = frag2(pbp + ln * 40 + hh * 16, pbp + ln * 40 + hh * 16 + 8);
        #pragma unroll
        for (int nt = 0; nt < 4; nt++) {
            v16bf vf = frag1(&sV[buf][(nt * 16 + ln) * 40 + hh * 16]);
            o[nt] = wmma_bf16(pf, vf, o[nt]);
        }
        buf ^= 1;
    }

    // write O / l  ->  attn[b*T + t][h*64 + hd]  (bf16)
    const int bb = bh >> 4, h = bh & 15;
    #pragma unroll
    for (int v = 0; v < 8; v++) {
        int trow = qr0 + v + 8 * hh;
        float inv = 1.0f / lst[v];
        size_t rowbase = ((size_t)bb * DIM_T + trow) * DIM_D + h * DIM_HD;
        #pragma unroll
        for (int nt = 0; nt < 4; nt++)
            out[rowbase + nt * 16 + ln] = f2bf(o[nt][v] * inv);
    }
}

// ---------------------------------------------------------------------------
// Host-side orchestration
// ---------------------------------------------------------------------------
extern "C" void kernel_launch(void* const* d_in, const int* in_sizes, int n_in,
                              void* d_out, int out_size, void* d_ws, size_t ws_size,
                              hipStream_t stream) {
    const float* x      = (const float*)d_in[0];
    const float* w_qkv  = (const float*)d_in[1];
    const float* w_proj = (const float*)d_in[2];
    const float* ln1_w  = (const float*)d_in[3];
    const float* ln1_b  = (const float*)d_in[4];
    const float* ln2_w  = (const float*)d_in[5];
    const float* ln2_b  = (const float*)d_in[6];
    const float* fc1_w  = (const float*)d_in[7];
    const float* fc1_b  = (const float*)d_in[8];
    const float* fc2_w  = (const float*)d_in[9];
    const float* fc2_b  = (const float*)d_in[10];
    float* outp = (float*)d_out;

    // carve workspace (256B-aligned slices)
    size_t off = 0;
    auto us = [&](size_t n) {
        unsigned short* p = (unsigned short*)((char*)d_ws + off);
        off = (off + n * 2 + 255) & ~(size_t)255;
        return p;
    };
    auto fl = [&](size_t n) {
        float* p = (float*)((char*)d_ws + off);
        off = (off + n * 4 + 255) & ~(size_t)255;
        return p;
    };
    unsigned short* wqkvT = us((size_t)3 * DIM_D * DIM_D);   // [3072,1024]
    unsigned short* wprjT = us((size_t)DIM_D * DIM_D);       // [1024,1024]
    unsigned short* fc1T  = us((size_t)DIM_FF * DIM_D);      // [4096,1024]
    unsigned short* fc2T  = us((size_t)DIM_D * DIM_FF);      // [1024,4096]
    unsigned short* xn1   = us((size_t)DIM_M * DIM_D);
    unsigned short* qb    = us((size_t)DIM_M * DIM_D);
    unsigned short* kb    = us((size_t)DIM_M * DIM_D);
    unsigned short* vb    = us((size_t)DIM_M * DIM_D);
    unsigned short* attn  = us((size_t)DIM_M * DIM_D);
    unsigned short* xn2   = us((size_t)DIM_M * DIM_D);
    unsigned short* h1    = us((size_t)DIM_M * DIM_FF);
    float*          x2    = fl((size_t)DIM_M * DIM_D);

    dim3 tb(32, 8);
    // weight transpose+convert: in[K,N] -> out[N,K] bf16
    tconv_kernel<<<dim3(3 * DIM_D / 32, DIM_D / 32), tb, 0, stream>>>(w_qkv, wqkvT, DIM_D, 3 * DIM_D);
    tconv_kernel<<<dim3(DIM_D / 32, DIM_D / 32),     tb, 0, stream>>>(w_proj, wprjT, DIM_D, DIM_D);
    tconv_kernel<<<dim3(DIM_FF / 32, DIM_D / 32),    tb, 0, stream>>>(fc1_w, fc1T, DIM_D, DIM_FF);
    tconv_kernel<<<dim3(DIM_D / 32, DIM_FF / 32),    tb, 0, stream>>>(fc2_w, fc2T, DIM_FF, DIM_D);

    // LN1
    ln_kernel<<<DIM_M, 256, 0, stream>>>(x, ln1_w, ln1_b, xn1);

    // QKV projection + scatter (q scaled by 1/sqrt(HD))
    gemm_bf16_kernel<0><<<dim3(3 * DIM_D / 64, DIM_M / 128), 256, 0, stream>>>(
        xn1, wqkvT, nullptr, nullptr, nullptr, nullptr, qb, kb, vb,
        DIM_M, 3 * DIM_D, DIM_D);

    // causal flash attention
    attn_kernel<<<dim3(DIM_T / 64, DIM_B * DIM_NH), 128, 0, stream>>>(qb, kb, vb, attn);

    // output projection + residual -> x2 (f32)
    gemm_bf16_kernel<1><<<dim3(DIM_D / 64, DIM_M / 128), 256, 0, stream>>>(
        attn, wprjT, nullptr, x, x2, nullptr, nullptr, nullptr, nullptr,
        DIM_M, DIM_D, DIM_D);

    // LN2
    ln_kernel<<<DIM_M, 256, 0, stream>>>(x2, ln2_w, ln2_b, xn2);

    // FC1 + bias + exact GELU -> h1 (bf16)
    gemm_bf16_kernel<2><<<dim3(DIM_FF / 64, DIM_M / 128), 256, 0, stream>>>(
        xn2, fc1T, fc1_b, nullptr, nullptr, h1, nullptr, nullptr, nullptr,
        DIM_M, DIM_FF, DIM_D);

    // FC2 + bias + residual -> out (f32)
    gemm_bf16_kernel<3><<<dim3(DIM_D / 64, DIM_M / 128), 256, 0, stream>>>(
        h1, fc2T, fc2_b, x2, outp, nullptr, nullptr, nullptr, nullptr,
        DIM_M, DIM_D, DIM_FF);
}
